// MambaBlock_46127948759439
// MI455X (gfx1250) — matrix operand order repaired and checked by reading, not compiled, not run on hardware
//
#include <hip/hip_runtime.h>
#include <hip/hip_bf16.h>

// ---- problem constants (from reference) ----
#define BATCH   2
#define SEQ     2048
#define BL      (BATCH * SEQ)      // 4096 rows
#define DMODEL  768
#define DINNER  1536
#define DIN2    3072               // 2*DINNER
#define DTRANK  48
#define DTPAD   64                 // K=48 zero-padded to 64 (multiple of 32)
#define DSTATE  16
#define RN      (DTRANK + 2 * DSTATE)   // 80
#define DCONV   4

typedef __attribute__((ext_vector_type(16))) __bf16 v16bf;
typedef __attribute__((ext_vector_type(8)))  __bf16 v8bf;
typedef __attribute__((ext_vector_type(8)))  float  v8f;

// float -> bf16, round-to-nearest-even
__device__ __forceinline__ __bf16 f2bf(float f) {
    unsigned u = __builtin_bit_cast(unsigned, f);
    unsigned r = (u + 0x7FFFu + ((u >> 16) & 1u)) >> 16;
    unsigned short s = (unsigned short)r;
    return __builtin_bit_cast(__bf16, s);
}
__device__ __forceinline__ float silu_f(float x) { return x / (1.0f + __expf(-x)); }
__device__ __forceinline__ float softplus_f(float x) {
    return (x > 20.0f) ? x : log1pf(__expf(x));
}

// ---------------------------------------------------------------------------
// WMMA GEMM: C[M,N](f32) = act(A[M,K](bf16 row-major) @ B(+bias))
// B is supplied PRE-TRANSPOSED as BT[N][K] bf16 -> every fragment load is two
// contiguous 16-byte v8bf loads (global_load_b128), same as the A side.
// Wave macro-tile: (16*MT) x (16*NT); MT*NT accumulators, MT*NT wmma/k-step.
// Invariants: M%(16*MT)==0, N%(16*NT)==0, K%32==0 (dt GEMM zero-padded).
// Fragment layouts per CDNA5 ISA 7.12.2 (wave32):
//   A(16x32): lane m=l&15, h=l>>4; elems 0..7 -> K=k0+8h+i, 8..15 -> k0+16+8h+i
//   B(32x16): lane n=l&15, h=l>>4; elems 0..7 -> K=k0+8h+i, 8..15 -> k0+16+8h+i
//   C/D(16x16 f32): lane n=l&15; vgpr r -> row m=8h+r
// act: 0 none, 1 softplus(x + bias[col])
// ---------------------------------------------------------------------------
template<int MT, int NT, int ACT>
__global__ __launch_bounds__(256)
void wmma_gemm(const __bf16* __restrict__ A, const __bf16* __restrict__ BT,
               const float* __restrict__ bias, float* __restrict__ C,
               int M, int N, int K, int lda, int ldc) {
    const int ntiles = N / (16 * NT);
    const int total  = (M / (16 * MT)) * ntiles;
    const int wave   = threadIdx.x >> 5;
    const int gtile  = blockIdx.x * 8 + wave;
    if (gtile >= total) return;                 // wave-uniform; EXEC stays full

    const int mg = gtile / ntiles;
    const int ng = gtile - mg * ntiles;

    const int lane = threadIdx.x & 31;
    const int hh   = lane >> 4;
    const int mr   = lane & 15;

    const int row0  = mg * 16 * MT + mr;
    const int ncol0 = ng * 16 * NT;

    v8f acc[MT][NT];
    #pragma unroll
    for (int mi = 0; mi < MT; ++mi)
        #pragma unroll
        for (int t = 0; t < NT; ++t) acc[mi][t] = (v8f){};

    for (int k0 = 0; k0 < K; k0 += 32) {
        v16bf af[MT];
        #pragma unroll
        for (int mi = 0; mi < MT; ++mi) {
            const __bf16* ap = A + (size_t)(row0 + 16 * mi) * lda + k0 + 8 * hh;
            v8bf alo = *(const v8bf*)(ap);       // 16B
            v8bf ahi = *(const v8bf*)(ap + 16);  // 16B
            af[mi] = __builtin_shufflevector(alo, ahi,
                        0,1,2,3,4,5,6,7,8,9,10,11,12,13,14,15);
        }
        v16bf bfv[NT];
        #pragma unroll
        for (int t = 0; t < NT; ++t) {
            const __bf16* bp =
                BT + (size_t)(ncol0 + 16 * t + mr) * K + k0 + 8 * hh;
            v8bf b0 = *(const v8bf*)(bp);        // 16B
            v8bf b1 = *(const v8bf*)(bp + 16);   // 16B
            bfv[t] = __builtin_shufflevector(b0, b1,
                        0,1,2,3,4,5,6,7,8,9,10,11,12,13,14,15);
        }
        #pragma unroll
        for (int mi = 0; mi < MT; ++mi)
            #pragma unroll
            for (int t = 0; t < NT; ++t)
                acc[mi][t] = __builtin_amdgcn_wmma_f32_16x16x32_bf16(
                                 false, af[mi], false, bfv[t],
                                 (short)0, acc[mi][t], false, false);
    }

    #pragma unroll
    for (int mi = 0; mi < MT; ++mi)
        #pragma unroll
        for (int t = 0; t < NT; ++t) {
            const int col = ncol0 + 16 * t + mr;
            const float bv = (ACT == 1) ? bias[col] : 0.0f;
            #pragma unroll
            for (int r = 0; r < 8; ++r) {
                const int row = row0 - mr + 16 * mi + 8 * hh + r;
                float v = acc[mi][t][r];
                if (ACT == 1) v = softplus_f(v + bv);
                C[(size_t)row * ldc + col] = v;
            }
        }
}

// ---------------------------------------------------------------------------
// f32 -> bf16 bulk convert (row-major A operands)
// ---------------------------------------------------------------------------
__global__ __launch_bounds__(256)
void cvt_bf16_kernel(const float* __restrict__ s, __bf16* __restrict__ d, size_t n) {
    size_t i = (size_t)blockIdx.x * blockDim.x + threadIdx.x;
    if (i < n) d[i] = f2bf(s[i]);
}

// f32 [R][Cn] row-major -> bf16 [Cn][R] (transposed, for B operands)
__global__ __launch_bounds__(256)
void transpose_bf16_kernel(const float* __restrict__ s, __bf16* __restrict__ d,
                           int R, int Cn) {
    size_t i = (size_t)blockIdx.x * blockDim.x + threadIdx.x;
    if (i >= (size_t)R * Cn) return;
    int r = (int)(i / Cn), c = (int)(i - (size_t)r * Cn);
    d[(size_t)c * R + r] = f2bf(s[i]);
}

// W_dt [48][1536] -> transposed + K-padded bf16 [1536][64]
__global__ __launch_bounds__(256)
void pad_wdt_t_kernel(const float* __restrict__ W_dt, __bf16* __restrict__ d) {
    int i = blockIdx.x * blockDim.x + threadIdx.x;
    if (i >= DINNER * DTPAD) return;
    int c = i / DTPAD, r = i - c * DTPAD;      // output [c][r], coalesced writes
    d[i] = (r < DTRANK) ? f2bf(W_dt[r * DINNER + c]) : f2bf(0.0f);
}

// dt_raw = ssm[:, :48] -> row-major K-padded bf16 [BL][64]
__global__ __launch_bounds__(256)
void pad_dtraw_kernel(const float* __restrict__ ssm, __bf16* __restrict__ d) {
    size_t i = (size_t)blockIdx.x * blockDim.x + threadIdx.x;
    if (i >= (size_t)BL * DTPAD) return;
    size_t row = i / DTPAD;
    int   c    = (int)(i - row * DTPAD);
    d[i] = (c < DTRANK) ? f2bf(ssm[row * RN + c]) : f2bf(0.0f);
}

// ---------------------------------------------------------------------------
// Depthwise causal conv (taps=4, left pad 3) + SiLU over u = xz[:, :DINNER]
// Writes f32 (for the scan) and bf16 (GEMM A operand) copies.
// ---------------------------------------------------------------------------
__global__ __launch_bounds__(256)
void conv_silu_kernel(const float* __restrict__ xz, const float* __restrict__ cw,
                      const float* __restrict__ cb, float* __restrict__ ucv,
                      __bf16* __restrict__ ucv_bf) {
    size_t idx = (size_t)blockIdx.x * blockDim.x + threadIdx.x;
    if (idx >= (size_t)BL * DINNER) return;
    int e  = (int)(idx % DINNER);
    int bl = (int)(idx / DINNER);
    int l  = bl % SEQ;
    int b  = bl / SEQ;

    float acc = cb[e];
    #pragma unroll
    for (int j = 0; j < DCONV; ++j) {
        int ll = l - (DCONV - 1) + j;
        if (ll >= 0)
            acc += cw[e * DCONV + j] * xz[((size_t)b * SEQ + ll) * DIN2 + e];
    }
    float v = silu_f(acc);
    ucv[idx]    = v;
    ucv_bf[idx] = f2bf(v);
}

// ---------------------------------------------------------------------------
// Sequential SSM recurrence, fused with D-skip and SiLU gate.
// One thread per (batch, channel): h[16] register-resident, 2048 steps.
// ---------------------------------------------------------------------------
__global__ __launch_bounds__(256)
void scan_kernel(const float* __restrict__ xz, const float* __restrict__ ucv,
                 const float* __restrict__ ssm, const float* __restrict__ dt,
                 const float* __restrict__ A_log, const float* __restrict__ Dp,
                 __bf16* __restrict__ yg_bf) {
    int idx = blockIdx.x * blockDim.x + threadIdx.x;
    if (idx >= BATCH * DINNER) return;
    int b = idx / DINNER;
    int e = idx % DINNER;

    float a[DSTATE], h[DSTATE];
    #pragma unroll
    for (int n = 0; n < DSTATE; ++n) {
        a[n] = -__expf(A_log[e * DSTATE + n]);
        h[n] = 0.0f;
    }
    const float dpe = Dp[e];

    for (int l = 0; l < SEQ; ++l) {
        const size_t row = (size_t)b * SEQ + l;
        const float dtv = dt[row * DINNER + e];
        const float uv  = ucv[row * DINNER + e];
        const float* s  = ssm + row * RN;
        float y = 0.0f;
        #pragma unroll
        for (int n = 0; n < DSTATE; ++n) {
            float dA  = __expf(dtv * a[n]);
            float dBu = dtv * s[DTRANK + n] * uv;
            h[n] = fmaf(dA, h[n], dBu);
            y    = fmaf(h[n], s[DTRANK + DSTATE + n], y);
        }
        const float res = xz[row * DIN2 + DINNER + e];
        yg_bf[row * DINNER + e] = f2bf((y + uv * dpe) * silu_f(res));
    }
}

// ---------------------------------------------------------------------------
extern "C" void kernel_launch(void* const* d_in, const int* in_sizes, int n_in,
                              void* d_out, int out_size, void* d_ws, size_t ws_size,
                              hipStream_t stream) {
    const float* x      = (const float*)d_in[0];
    const float* W_in   = (const float*)d_in[1];
    const float* conv_w = (const float*)d_in[2];
    const float* conv_b = (const float*)d_in[3];
    const float* W_x    = (const float*)d_in[4];
    const float* W_dt   = (const float*)d_in[5];
    const float* b_dt   = (const float*)d_in[6];
    const float* A_log  = (const float*)d_in[7];
    const float* Dp     = (const float*)d_in[8];
    const float* W_out  = (const float*)d_in[9];
    float* out = (float*)d_out;

    // ---- workspace carve-up: f32 region then bf16 region (~141 MB) ----
    float* fws = (float*)d_ws;
    float* xz  = fws;                               // [BL,3072] f32
    float* ucv = xz  + (size_t)BL * DIN2;           // [BL,1536] f32
    float* ssm = ucv + (size_t)BL * DINNER;         // [BL,80]   f32
    float* dt  = ssm + (size_t)BL * RN;             // [BL,1536] f32
    __bf16* bws    = (__bf16*)(dt + (size_t)BL * DINNER);
    __bf16* x_bf   = bws;                           // [BL,768]        A
    __bf16* ucv_bf = x_bf   + (size_t)BL * DMODEL;  // [BL,1536]       A
    __bf16* dtp_bf = ucv_bf + (size_t)BL * DINNER;  // [BL,64]         A
    __bf16* yg_bf  = dtp_bf + (size_t)BL * DTPAD;   // [BL,1536]       A
    __bf16* Win_bt = yg_bf  + (size_t)BL * DINNER;  // [3072,768]  B^T
    __bf16* Wx_bt  = Win_bt + (size_t)DIN2 * DMODEL;// [80,1536]   B^T
    __bf16* Wdt_bt = Wx_bt  + (size_t)RN * DINNER;  // [1536,64]   B^T
    __bf16* Wout_bt= Wdt_bt + (size_t)DINNER * DTPAD;//[768,1536]  B^T

    auto blocks = [](int M, int N, int mt, int nt) {
        return ((M / (16 * mt)) * (N / (16 * nt)) + 7) / 8;
    };

    // 0) operand conversions / transposes (bandwidth-trivial)
    cvt_bf16_kernel<<<(int)(((size_t)BL * DMODEL + 255) / 256), 256, 0, stream>>>(
        x, x_bf, (size_t)BL * DMODEL);
    transpose_bf16_kernel<<<(int)(((size_t)DMODEL * DIN2 + 255) / 256), 256, 0, stream>>>(
        W_in, Win_bt, DMODEL, DIN2);
    transpose_bf16_kernel<<<(int)(((size_t)DINNER * RN + 255) / 256), 256, 0, stream>>>(
        W_x, Wx_bt, DINNER, RN);
    transpose_bf16_kernel<<<(int)(((size_t)DINNER * DMODEL + 255) / 256), 256, 0, stream>>>(
        W_out, Wout_bt, DINNER, DMODEL);
    pad_wdt_t_kernel<<<(DINNER * DTPAD + 255) / 256, 256, 0, stream>>>(W_dt, Wdt_bt);

    // 1) xz = x @ W_in                     [4096,768]x[768,3072]
    wmma_gemm<2, 4, 0><<<blocks(BL, DIN2, 2, 4), 256, 0, stream>>>(
        x_bf, Win_bt, nullptr, xz, BL, DIN2, DMODEL, DMODEL, DIN2);

    // 2) u = silu(causal_dwconv(xz[:, :E]))
    conv_silu_kernel<<<(int)(((size_t)BL * DINNER + 255) / 256), 256, 0, stream>>>(
        xz, conv_w, conv_b, ucv, ucv_bf);

    // 3) ssm = u @ W_x                     [4096,1536]x[1536,80]
    wmma_gemm<2, 1, 0><<<blocks(BL, RN, 2, 1), 256, 0, stream>>>(
        ucv_bf, Wx_bt, nullptr, ssm, BL, RN, DINNER, DINNER, RN);

    // 4) dt = softplus(dt_raw @ W_dt + b)  [4096,64pad]x[64pad,1536]
    pad_dtraw_kernel<<<(int)(((size_t)BL * DTPAD + 255) / 256), 256, 0, stream>>>(
        ssm, dtp_bf);
    wmma_gemm<2, 4, 1><<<blocks(BL, DINNER, 2, 4), 256, 0, stream>>>(
        dtp_bf, Wdt_bt, b_dt, dt, BL, DINNER, DTPAD, DTPAD, DINNER);

    // 5) sequential SSM scan + D skip + SiLU gate -> bf16
    scan_kernel<<<(BATCH * DINNER + 255) / 256, 256, 0, stream>>>(
        xz, ucv, ssm, dt, A_log, Dp, yg_bf);

    // 6) out = yg @ W_out                  [4096,1536]x[1536,768]
    wmma_gemm<2, 4, 0><<<blocks(BL, DMODEL, 2, 4), 256, 0, stream>>>(
        yg_bf, Wout_bt, nullptr, out, BL, DMODEL, DINNER, DINNER, DMODEL);
}
